// VectorQuantizer2_19885698580820
// MI455X (gfx1250) — matrix operand order repaired and checked by reading, not compiled, not run on hardware
//
#include <hip/hip_runtime.h>

// ---------------------------------------------------------------------------
// VQ-VAE vector quantizer for MI455X (gfx1250, wave32).
//   d_in[0] = z      : (32,256,32,32) f32  -> rows = b*1024 + h*32 + w, K = c
//   d_in[1] = emb_w  : (1024,256)     f32
//   d_out (flat f32, concat in return order):
//     [0)            z_q_out (32,256,32,32)           8,388,608
//     [8388608)      loss                             1
//     [8388609)      perplexity                       1
//     [8388610)      min_encodings (32768,1024)       33,554,432
//     [41943042)     min_encoding_indices (32768,1)   32,768
// Matmul path: bf16 V_WMMA_F32_16X16X32_BF16, f32 accumulate. argmin only
// needs the relative ordering of (||e||^2 - 2 z.e); z_q / loss use f32 data.
// ---------------------------------------------------------------------------

#define N_ROWS   32768
#define K_DIM    256
#define N_CODES  1024
#define HW_SZ    1024
#define TILE_M   64
#define PITCH    264      // 256 + 8 bf16 pad -> breaks LDS bank alignment

// workspace layout (bytes)
#define WS_EMBT   0              // 1024*256 bf16 = 524288
#define WS_ENORM  524288         // 1024 f32
#define WS_COUNTS 528384         // 1024 u32
#define WS_IDX    532480         // 32768 i32
#define WS_PART   663552         // 1024 f32 loss partials

typedef __attribute__((ext_vector_type(16))) __bf16 v16bf;
typedef __attribute__((ext_vector_type(8)))  float  v8f;

__device__ __forceinline__ unsigned short f32_to_bf16_rne(float f) {
    unsigned u = __builtin_bit_cast(unsigned, f);
    return (unsigned short)((u + 0x7FFFu + ((u >> 16) & 1u)) >> 16);
}

// A-fragment: two 16B chunks, adjacent at the call site (16 consecutive K, as B).
__device__ __forceinline__ v16bf load_frag_split(const unsigned short* p0,
                                                 const unsigned short* p1) {
    v16bf r;
    __builtin_memcpy(&r, p0, 16);
    __builtin_memcpy(reinterpret_cast<char*>(&r) + 16, p1, 16);
    return r;
}

// B-fragment: 16 consecutive K values (contiguous k order, same as A).
__device__ __forceinline__ v16bf load_frag_contig(const unsigned short* p) {
    v16bf r;
    __builtin_memcpy(&r, p, 32);
    return r;
}

// ---------------------------------------------------------------- init -----
__global__ __launch_bounds__(1024) void vq_init(unsigned* __restrict__ counts) {
    counts[threadIdx.x] = 0u;
}

// ------------------------------------------------- emb -> bf16 + ||e||^2 ---
__global__ __launch_bounds__(256) void vq_prep_emb(const float* __restrict__ emb,
                                                   unsigned short* __restrict__ embT,
                                                   float* __restrict__ enorm) {
    const int n = blockIdx.x;          // one code per block
    const int t = threadIdx.x;         // one element per thread
    float v = emb[n * K_DIM + t];
    embT[n * K_DIM + t] = f32_to_bf16_rne(v);
    float s = v * v;
    #pragma unroll
    for (int off = 16; off >= 1; off >>= 1) s += __shfl_xor(s, off, 32);
    __shared__ float sh[8];
    if ((t & 31) == 0) sh[t >> 5] = s;
    __syncthreads();
    if (t == 0) {
        float tot = 0.f;
        #pragma unroll
        for (int i = 0; i < 8; ++i) tot += sh[i];
        enorm[n] = tot;
    }
}

// --------------------------------------------- WMMA distance + argmin ------
__global__ __launch_bounds__(256) void vq_argmin(const float* __restrict__ z,
                                                 const unsigned short* __restrict__ embT,
                                                 const float* __restrict__ enorm,
                                                 int* __restrict__ idxOut,
                                                 float* __restrict__ idxOutF,
                                                 unsigned* __restrict__ counts) {
    __shared__ unsigned short ldsZ[TILE_M * PITCH];   // 33,792 B, z tile in bf16
    __shared__ float ldsEn[N_CODES];                  //  4,096 B
    __shared__ float ldsBestD[2 * TILE_M];
    __shared__ int   ldsBestN[2 * TILE_M];

    const int t    = threadIdx.x;
    const int row0 = blockIdx.x * TILE_M;             // 512 blocks
    const int b    = row0 >> 10;                      // rows in a block share b
    const int hw0  = row0 & 1023;

    for (int i = t; i < N_CODES; i += 256) ldsEn[i] = enorm[i];

    // Stage z tile into LDS as bf16, doing the NCHW->row-major(K) transpose.
    {
        const int j  = t & 63;       // row within tile (consecutive hw)
        const int cq = t >> 6;       // 0..3
        for (int c = cq; c < K_DIM; c += 4) {
            float v = z[(size_t)(b * K_DIM + c) * HW_SZ + hw0 + j];
            ldsZ[j * PITCH + c] = f32_to_bf16_rne(v);
        }
    }
    __syncthreads();

    const int wave  = t >> 5;
    const int lane  = t & 31;
    const int kh    = lane >> 4;      // half-wave
    const int lm    = lane & 15;
    const int mSub  = wave & 3;       // 4 sub-tiles of 16 rows
    const int nHalf = wave >> 2;      // waves split the 64 N-tiles in half

    // Hoist the wave's entire A (16 rows x 256 K) into registers: 8 x v16bf.
    // A layout (matches B): lane lm = row; elems 0..15 -> K = ks*32 + kh*16 + e.
    const unsigned short* zrow = &ldsZ[(mSub * 16 + lm) * PITCH];
    v16bf afrag[8];
    #pragma unroll
    for (int ks = 0; ks < 8; ++ks) {
        const unsigned short* p = zrow + ks * 32 + kh * 16;
        afrag[ks] = load_frag_split(p, p + 8);
    }

    float bestd[8];
    int   bestn[8];
    #pragma unroll
    for (int r = 0; r < 8; ++r) { bestd[r] = 3.4e38f; bestn[r] = 0; }

    const int ntEnd = nHalf * 32 + 32;
    for (int nt = nHalf * 32; nt < ntEnd; ++nt) {
        const int n = nt * 16 + lm;            // this lane's column / code id
        v8f acc = {};
        // B layout (matches A): lane lm = column n; elems 0..15 -> K = ks*32 + kh*16 + e
        const unsigned short* ecol = embT + (size_t)n * K_DIM + kh * 16;
        #pragma unroll
        for (int ks = 0; ks < 8; ++ks) {
            v16bf bfrag = load_frag_contig(ecol + ks * 32);
            acc = __builtin_amdgcn_wmma_f32_16x16x32_bf16(
                false, afrag[ks], false, bfrag, (short)0, acc, false, false);
        }
        const float en = ldsEn[n];
        #pragma unroll
        for (int r = 0; r < 8; ++r) {
            float d = en - 2.0f * acc[r];      // ||z||^2 constant -> dropped
            if (d < bestd[r]) { bestd[r] = d; bestn[r] = n; }  // first-min tie
        }
    }

    // Cross-lane argmin: C/D layout -> lanes 0..15 hold rows kh*8+r for kh=0,
    // lanes 16..31 for kh=1; reduce over the 16 columns (prefer smaller n).
    #pragma unroll
    for (int r = 0; r < 8; ++r) {
        float d = bestd[r];
        int   n = bestn[r];
        #pragma unroll
        for (int off = 8; off >= 1; off >>= 1) {
            float od = __shfl_xor(d, off, 16);
            int   on = __shfl_xor(n, off, 16);
            if (od < d || (od == d && on < n)) { d = od; n = on; }
        }
        if (lm == 0) {
            int tr = mSub * 16 + kh * 8 + r;
            ldsBestD[nHalf * TILE_M + tr] = d;
            ldsBestN[nHalf * TILE_M + tr] = n;
        }
    }
    __syncthreads();

    if (t < TILE_M) {   // combine the two N halves (n0 < 512 <= n1: tie->n0)
        float d0 = ldsBestD[t];          int n0 = ldsBestN[t];
        float d1 = ldsBestD[TILE_M + t]; int n1 = ldsBestN[TILE_M + t];
        if (d1 < d0) { d0 = d1; n0 = n1; }
        const int grow = row0 + t;
        idxOut[grow]  = n0;
        idxOutF[grow] = (float)n0;
        atomicAdd(&counts[n0], 1u);
    }
}

// ----------------------------------- z_q gather + deterministic loss -------
__global__ __launch_bounds__(256) void vq_zq_loss(const float* __restrict__ z,
                                                  const float* __restrict__ emb,
                                                  const int* __restrict__ idx,
                                                  float* __restrict__ zq_out,
                                                  float* __restrict__ partials) {
    const unsigned total  = (unsigned)N_ROWS * K_DIM;   // 8,388,608 (NCHW flat)
    const unsigned stride = gridDim.x * 256u;
    float acc = 0.f;
    for (unsigned i = blockIdx.x * 256u + threadIdx.x; i < total; i += stride) {
        const unsigned hw = i & 1023u;
        const unsigned bc = i >> 10;
        const int c   = (int)(bc & 255u);
        const int bb  = (int)(bc >> 8);
        const int row = (bb << 10) + (int)hw;
        const float q = emb[(size_t)idx[row] * K_DIM + c];   // 1 MB, L2-hot
        const float d = q - z[i];
        zq_out[i] = q;     // forward value of z_q_st == z_q
        acc += d * d;
    }
    #pragma unroll
    for (int off = 16; off >= 1; off >>= 1) acc += __shfl_xor(acc, off, 32);
    __shared__ float sh[8];
    if ((threadIdx.x & 31) == 0) sh[threadIdx.x >> 5] = acc;
    __syncthreads();
    if (threadIdx.x == 0) {
        float s = 0.f;
        #pragma unroll
        for (int i = 0; i < 8; ++i) s += sh[i];
        partials[blockIdx.x] = s;   // fixed-order final sum -> deterministic
    }
}

// ----------------------------------------------- one-hot (single pass) -----
__global__ __launch_bounds__(256) void vq_onehot(const int* __restrict__ idx,
                                                 float* __restrict__ onehot) {
    const int row = blockIdx.x;
    const int k   = idx[row];
    const int c0  = threadIdx.x * 4;
    float2* p = reinterpret_cast<float2*>(onehot + (size_t)row * N_CODES + c0);
    p[0] = make_float2(c0     == k ? 1.f : 0.f, c0 + 1 == k ? 1.f : 0.f);
    p[1] = make_float2(c0 + 2 == k ? 1.f : 0.f, c0 + 3 == k ? 1.f : 0.f);
}

// ----------------------------------------------- loss + perplexity ---------
__global__ __launch_bounds__(1024) void vq_finalize(const unsigned* __restrict__ counts,
                                                    const float* __restrict__ partials,
                                                    float* __restrict__ lossOut,
                                                    float* __restrict__ perpOut) {
    const int t = threadIdx.x;
    float p = (float)counts[t] * (1.0f / (float)N_ROWS);
    float term = p * logf(p + 1e-10f);
    #pragma unroll
    for (int off = 16; off >= 1; off >>= 1) term += __shfl_xor(term, off, 32);
    __shared__ float sh[32];
    if ((t & 31) == 0) sh[t >> 5] = term;
    __syncthreads();
    if (t == 0) {
        float s = 0.f;
        #pragma unroll
        for (int i = 0; i < 32; ++i) s += sh[i];
        *perpOut = expf(-s);
        float ls = 0.f;
        for (int i = 0; i < 1024; ++i) ls += partials[i];
        *lossOut = 1.25f * ls * (1.0f / 8388608.0f);   // (1+BETA)*mean
    }
}

// ---------------------------------------------------------------------------
extern "C" void kernel_launch(void* const* d_in, const int* in_sizes, int n_in,
                              void* d_out, int out_size, void* d_ws, size_t ws_size,
                              hipStream_t stream) {
    const float* z   = (const float*)d_in[0];
    const float* emb = (const float*)d_in[1];

    float* out     = (float*)d_out;
    float* zq_out  = out;                        // 8,388,608
    float* lossOut = out + 8388608;
    float* perpOut = out + 8388609;
    float* onehot  = out + 8388610;              // 33,554,432 (8B aligned)
    float* idxOutF = out + 41943042;             // 32,768

    char* ws = (char*)d_ws;
    unsigned short* embT    = (unsigned short*)(ws + WS_EMBT);
    float*          enorm   = (float*)(ws + WS_ENORM);
    unsigned*       counts  = (unsigned*)(ws + WS_COUNTS);
    int*            idxI    = (int*)(ws + WS_IDX);
    float*          partials= (float*)(ws + WS_PART);

    vq_init    <<<1,                1024, 0, stream>>>(counts);
    vq_prep_emb<<<N_CODES,           256, 0, stream>>>(emb, embT, enorm);
    vq_argmin  <<<N_ROWS / TILE_M,   256, 0, stream>>>(z, embT, enorm,
                                                       idxI, idxOutF, counts);
    vq_zq_loss <<<1024,              256, 0, stream>>>(z, emb, idxI, zq_out, partials);
    vq_onehot  <<<N_ROWS,            256, 0, stream>>>(idxI, onehot);
    vq_finalize<<<1,                1024, 0, stream>>>(counts, partials, lossOut, perpOut);
}
